// SkippingDiffusionLayer_9955734192271
// MI455X (gfx1250) — compile-verified, matrix operand order checked by reading, not run on hardware
//
#include <hip/hip_runtime.h>
#include <hip/hip_bf16.h>

// Problem constants from the reference
#define BB   16
#define NN   2048
#define DD   256
#define ROWS (BB * NN)          // 32768
#define ELEMS ((size_t)BB * NN * DD)

typedef __attribute__((ext_vector_type(16))) __bf16 v16bf;
typedef __attribute__((ext_vector_type(8)))  float  v8f;
typedef __attribute__((ext_vector_type(4)))  unsigned int v4u;
typedef __attribute__((ext_vector_type(8)))  int  v8i;
typedef __attribute__((ext_vector_type(4)))  int  v4i;

// LDS tile: 16 rows x 256 bf16 (512B) + 16B pad per row -> 528B row stride
#define LDS_ROW_BF 264                  // 528 bytes / 2
#define LDS_TILE_BF (16 * LDS_ROW_BF)   // 4224 bf16 = 8448 B

// ---------------------------------------------------------------------------
// K0: f32 -> bf16 conversion of both inputs (feeds the WMMA GEMM)
// ---------------------------------------------------------------------------
__global__ __launch_bounds__(256) void convert_kernel(const float* __restrict__ L,
                                                      const float* __restrict__ H,
                                                      __bf16* __restrict__ Lb,
                                                      __bf16* __restrict__ Hb) {
    size_t i = (size_t)blockIdx.x * blockDim.x + threadIdx.x;
    if (i < ELEMS) {
        Lb[i] = (__bf16)L[i];
        Hb[i] = (__bf16)H[i];
    }
}

// ---------------------------------------------------------------------------
// K1: row L2 norms + reciprocals
// ---------------------------------------------------------------------------
__global__ __launch_bounds__(256) void norm_kernel(const float* __restrict__ L,
                                                   const float* __restrict__ H,
                                                   float* __restrict__ nl,
                                                   float* __restrict__ nh,
                                                   float* __restrict__ rnl,
                                                   float* __restrict__ rnh) {
    __shared__ float red[256];
    int rb = blockIdx.x;
    const float* src;
    float *dst, *rdst;
    if (rb < ROWS) { src = L + (size_t)rb * DD; dst = nl + rb; rdst = rnl + rb; }
    else { int r2 = rb - ROWS; src = H + (size_t)r2 * DD; dst = nh + r2; rdst = rnh + r2; }
    float x = src[threadIdx.x];
    red[threadIdx.x] = x * x;
    __syncthreads();
    for (int off = 128; off > 0; off >>= 1) {
        if (threadIdx.x < off) red[threadIdx.x] += red[threadIdx.x + off];
        __syncthreads();
    }
    if (threadIdx.x == 0) {
        float nrm = sqrtf(red[0]);
        *dst  = nrm;
        *rdst = 1.0f / fmaxf(nrm, 1e-6f);
    }
}

// ---------------------------------------------------------------------------
// 16-bit WMMA operand loader (wave32 layout): elements 0..7 = K koff..koff+7,
// 8..15 = K koff+16..koff+23, koff = 8*(lane>=16).  Works for A (lane=M) and
// B (lane=N).  rowStride in __bf16 units.
// ---------------------------------------------------------------------------
static __device__ __forceinline__ v16bf load_tile(const __bf16* tileBase, int lane,
                                                  int k0, int rowStride) {
    int r    = lane & 15;
    int koff = (lane >> 4) << 3;
    const __bf16* p = tileBase + (size_t)r * rowStride + k0 + koff;
    union { v16bf v; uint4 q[2]; } u;
    u.q[0] = *(const uint4*)(p);
    u.q[1] = *(const uint4*)(p + 16);
    return u.v;
}

// Deterministic "gumbel" noise: hash(b,n,m) -> u in (0,1) -> -log(-log(u))
static __device__ __forceinline__ float gumbel_of(int b, int n, int m) {
    unsigned x = ((unsigned)b << 22) | ((unsigned)n << 11) | (unsigned)m;
    x ^= x >> 16; x *= 0x7feb352dU;
    x ^= x >> 15; x *= 0x846ca68bU;
    x ^= x >> 16;
    float u = fmaxf((float)(x >> 8) * (1.0f / 16777216.0f), 1e-10f);
    return -__logf(-__logf(u));
}

// ---------------------------------------------------------------------------
// TDM: stage one 16x256 bf16 A-tile (row stride 256 elems in memory) into LDS
// with 16B padding every 512B (pad_interval=6 -> 128 DW, pad_amount=3 -> 4 DW)
// D# layout per CDNA5 ISA ch.8.  Toolchain uses the 6-arg builtin form.
// ---------------------------------------------------------------------------
static __device__ __forceinline__ void tdm_stage_tile(const __bf16* gsrc,
                                                      unsigned lds_off) {
    unsigned long long ga = (unsigned long long)(uintptr_t)gsrc;
    v4u g0;
    g0.x = 1u;                                       // count=1 valid descriptor
    g0.y = lds_off;                                  // lds_addr (bytes)
    g0.z = (unsigned)(ga & 0xFFFFFFFFull);           // global_addr[31:0]
    g0.w = (unsigned)((ga >> 32) & 0x1FFFFFFull)     // global_addr[56:32]
         | (2u << 30);                               // type=2 ("image")
    v8i g1;
    g1[0] = (int)((1u << 16)                         // data_size=1 (2 bytes)
         | (1u << 20)                                // pad_enable
         | (6u << 22)                                // pad_interval: 128 DWORDs
         | (3u << 25));                              // pad_amount:   4 DWORDs
    g1[1] = (int)(((unsigned)DD & 0xFFFFu) << 16);   // tensor_dim0[15:0]=256
    g1[2] = (int)(((unsigned)NN & 0xFFFFu) << 16);   // dim0[31:16]=0 | tensor_dim1[15:0]=2048
    g1[3] = (int)(((unsigned)DD & 0xFFFFu) << 16);   // dim1[31:16]=0 | tile_dim0=256
    g1[4] = 16;                                      // tile_dim1=16 | tile_dim2=0
    g1[5] = DD;                                      // tensor_dim0_stride[31:0]=256
    g1[6] = 0;                                       // stride[47:32]=0 | dim1_stride lo
    g1[7] = 0;
    v4i zero4 = {0, 0, 0, 0};
    v8i zero8 = {0, 0, 0, 0, 0, 0, 0, 0};
    __builtin_amdgcn_tensor_load_to_lds(g0, g1, zero4, zero4, zero8, 0);
}

// ---------------------------------------------------------------------------
// K2: fused  sim = (LLF·HLFᵀ)·rnl·rnh/16  -> column softmax (over n)
//     -> argmax over n of (probs + gumbel).  Flash-style two-pass.
// Block = 128 threads = 4 waves = 4 m-tiles of one batch; A-tiles staged to
// LDS by the Tensor Data Mover (wave 0), double-buffered.  A tile preloaded
// into registers so the 8 WMMAs issue back-to-back (no per-step dscnt stall).
// ---------------------------------------------------------------------------
__global__ __launch_bounds__(128) void sim_argmax_kernel(const __bf16* __restrict__ Lb,
                                                         const __bf16* __restrict__ Hb,
                                                         const float* __restrict__ rnl,
                                                         const float* __restrict__ rnh,
                                                         int* __restrict__ idxOut) {
    __shared__ __align__(16) __bf16 atile[2][LDS_TILE_BF];

    const int wave = threadIdx.x >> 5;
    const int lane = threadIdx.x & 31;
    const int bid  = blockIdx.x;            // 0..511
    const int b    = bid >> 5;              // 32 blocks per batch
    const int mt   = (bid & 31) * 4 + wave; // m-tile 0..127
    const int m_base = mt * 16;
    const int r  = lane & 15;
    const int hi = lane >> 4;

    const __bf16* Abase = Lb + (size_t)b * NN * DD;
    const __bf16* Bbase = Hb + ((size_t)b * NN + m_base) * DD;
    const float*  rnlb  = rnl + (size_t)b * NN;

    const unsigned lds0 = (unsigned)(uintptr_t)(&atile[0][0]);
    const unsigned lds1 = (unsigned)(uintptr_t)(&atile[1][0]);

    // Hoist B (this wave's 16 HLF rows, full K=256): 8 x v16bf = 64 VGPRs
    v16bf breg[8];
#pragma unroll
    for (int kk = 0; kk < 8; ++kk) breg[kk] = load_tile(Bbase, lane, kk * 32, DD);

    // per-column scale: 1/(nl*nh)/sqrt(256)
    const float colScale = rnh[(size_t)b * NN + m_base + r] * 0.0625f;

    // ---- prologue: stage tile 0 --------------------------------------------
    if (threadIdx.x < 32) {
        tdm_stage_tile(Abase, lds0);
        __builtin_amdgcn_s_wait_tensorcnt(0);
    }
    __syncthreads();

    // -------- pass 1: online column softmax stats (max M, sum S) -----------
    float M = -3.0e38f, S = 0.0f;
    for (int nb = 0, ib = 0; nb < NN; nb += 16, ib ^= 1) {
        if (threadIdx.x < 32 && nb + 16 < NN)
            tdm_stage_tile(Abase + (size_t)(nb + 16) * DD, ib ? lds0 : lds1);
        const __bf16* Ab = atile[ib];
        // preload whole A tile -> all 16 ds_load_b128 in flight together
        v16bf areg[8];
#pragma unroll
        for (int kk = 0; kk < 8; ++kk) areg[kk] = load_tile(Ab, lane, kk * 32, LDS_ROW_BF);
        v8f c = {};
#pragma unroll
        for (int kk = 0; kk < 8; ++kk)
            c = __builtin_amdgcn_wmma_f32_16x16x32_bf16(false, areg[kk], false, breg[kk],
                                                        (short)0, c, false, false);
        float s[8];
#pragma unroll
        for (int j = 0; j < 8; ++j)
            s[j] = c[j] * (rnlb[nb + j + hi * 8] * colScale);
        float t0 = fmaxf(fmaxf(s[0], s[1]), fmaxf(s[2], s[3]));
        float t1 = fmaxf(fmaxf(s[4], s[5]), fmaxf(s[6], s[7]));
        float nM = fmaxf(M, fmaxf(t0, t1));
        float acc = 0.0f;
#pragma unroll
        for (int j = 0; j < 8; ++j) acc += __expf(s[j] - nM);
        S = S * __expf(M - nM) + acc;
        M = nM;
        if (threadIdx.x < 32) __builtin_amdgcn_s_wait_tensorcnt(0);
        __syncthreads();
    }
    {   // combine the two lane-halves of each column
        float oM = __shfl_xor(M, 16, 32);
        float oS = __shfl_xor(S, 16, 32);
        float nM = fmaxf(M, oM);
        S = S * __expf(M - nM) + oS * __expf(oM - nM);
        M = nM;
    }
    const float rZ = 1.0f / S;

    // -------- pass 2: argmax over n of  exp(s-M)/Z + gumbel -----------------
    if (threadIdx.x < 32) {
        tdm_stage_tile(Abase, lds0);
        __builtin_amdgcn_s_wait_tensorcnt(0);
    }
    __syncthreads();

    float best  = -3.0e38f;
    int   besti = 0;
    const int mcol = m_base + r;
    for (int nb = 0, ib = 0; nb < NN; nb += 16, ib ^= 1) {
        if (threadIdx.x < 32 && nb + 16 < NN)
            tdm_stage_tile(Abase + (size_t)(nb + 16) * DD, ib ? lds0 : lds1);
        const __bf16* Ab = atile[ib];
        v16bf areg[8];
#pragma unroll
        for (int kk = 0; kk < 8; ++kk) areg[kk] = load_tile(Ab, lane, kk * 32, LDS_ROW_BF);
        v8f c = {};
#pragma unroll
        for (int kk = 0; kk < 8; ++kk)
            c = __builtin_amdgcn_wmma_f32_16x16x32_bf16(false, areg[kk], false, breg[kk],
                                                        (short)0, c, false, false);
#pragma unroll
        for (int j = 0; j < 8; ++j) {
            int   n = nb + j + hi * 8;
            float s = c[j] * (rnlb[n] * colScale);
            float p = __expf(s - M) * rZ + gumbel_of(b, n, mcol);
            if (p > best) { best = p; besti = n; }
        }
        if (threadIdx.x < 32) __builtin_amdgcn_s_wait_tensorcnt(0);
        __syncthreads();
    }
    {
        float ov = __shfl_xor(best, 16, 32);
        int   oi = __shfl_xor(besti, 16, 32);
        if (ov > best || (ov == best && oi < besti)) { best = ov; besti = oi; }
    }
    if (lane < 16) idxOut[(size_t)b * NN + mcol] = besti;
}

// ---------------------------------------------------------------------------
// K3: per-batch counting sort of idx (values in [0,2048)) -> sl
// ---------------------------------------------------------------------------
__global__ __launch_bounds__(256) void sort_kernel(const int* __restrict__ idx,
                                                   int* __restrict__ sl) {
    __shared__ int cnt[NN];
    __shared__ int partial[256];
    const int b = blockIdx.x, t = threadIdx.x;
    for (int i = t; i < NN; i += 256) cnt[i] = 0;
    __syncthreads();
    for (int i = t; i < NN; i += 256) atomicAdd(&cnt[idx[b * NN + i]], 1);
    __syncthreads();
    int local[8], sum = 0;
#pragma unroll
    for (int k = 0; k < 8; ++k) { local[k] = sum; sum += cnt[t * 8 + k]; }
    partial[t] = sum;
    __syncthreads();
    int total = sum;
    for (int off = 1; off < 256; off <<= 1) {
        int v = (t >= off) ? partial[t - off] : 0;
        __syncthreads();
        partial[t] += v;
        __syncthreads();
    }
    int base = partial[t] - total;
#pragma unroll
    for (int k = 0; k < 8; ++k) {
        int v = t * 8 + k, o = base + local[k], c = cnt[v];
        for (int j = 0; j < c; ++j) sl[b * NN + o + j] = v;
    }
}

// ---------------------------------------------------------------------------
// K4a: w[b,n] = <HLF[b,sl[n]], LLF[b,n]> / max(nh[sl[n]]*nl[n], eps)
// ---------------------------------------------------------------------------
__global__ __launch_bounds__(128) void w_kernel(const float* __restrict__ L,
                                                const float* __restrict__ H,
                                                const float* __restrict__ nl,
                                                const float* __restrict__ nh,
                                                const int* __restrict__ sl,
                                                float* __restrict__ w) {
    int row  = blockIdx.x * 4 + (threadIdx.x >> 5);
    int lane = threadIdx.x & 31;
    int b = row >> 11, n = row & 2047;
    int j = sl[row];
    const float* lp = L + ((size_t)b * NN + n) * DD;
    const float* hp = H + ((size_t)b * NN + j) * DD;
    float s = 0.0f;
    for (int k = lane; k < DD; k += 32) s += lp[k] * hp[k];
#pragma unroll
    for (int off = 16; off > 0; off >>= 1) s += __shfl_xor(s, off, 32);
    if (lane == 0)
        w[row] = s / fmaxf(nh[b * NN + j] * nl[row], 1e-8f);
}

// ---------------------------------------------------------------------------
// K4b: noise[b,:] = softmax(w[b,:]/0.1)  (one block per batch)
// ---------------------------------------------------------------------------
__global__ __launch_bounds__(256) void noise_kernel(const float* __restrict__ w,
                                                    float* __restrict__ noise) {
    __shared__ float red[256];
    __shared__ float Msh, Zsh;
    const int b = blockIdx.x, t = threadIdx.x;
    float m = -3.0e38f;
    for (int i = t; i < NN; i += 256) m = fmaxf(m, w[b * NN + i] * 10.0f);
    red[t] = m;
    __syncthreads();
    for (int off = 128; off > 0; off >>= 1) {
        if (t < off) red[t] = fmaxf(red[t], red[t + off]);
        __syncthreads();
    }
    if (t == 0) Msh = red[0];
    __syncthreads();
    float M = Msh, s = 0.0f;
    for (int i = t; i < NN; i += 256) s += __expf(w[b * NN + i] * 10.0f - M);
    red[t] = s;
    __syncthreads();
    for (int off = 128; off > 0; off >>= 1) {
        if (t < off) red[t] += red[t + off];
        __syncthreads();
    }
    if (t == 0) Zsh = red[0];
    __syncthreads();
    float rZ = 1.0f / Zsh;
    for (int i = t; i < NN; i += 256)
        noise[b * NN + i] = __expf(w[b * NN + i] * 10.0f - M) * rZ;
}

// ---------------------------------------------------------------------------
// K5: out = HLF (vectorized copy)
// ---------------------------------------------------------------------------
__global__ __launch_bounds__(256) void copy_kernel(const float4* __restrict__ src,
                                                   float4* __restrict__ dst, size_t n4) {
    size_t i = (size_t)blockIdx.x * blockDim.x + threadIdx.x;
    if (i < n4) dst[i] = src[i];
}

// ---------------------------------------------------------------------------
// K6: scatter: out[b, sl[n], :] = HLF[b, sl[n], :] + LLF[b, n, :] * noise[b,n]
// (last occurrence of a duplicate index wins -> deterministic)
// ---------------------------------------------------------------------------
__global__ __launch_bounds__(256) void scatter_kernel(const float* __restrict__ L,
                                                      const float* __restrict__ H,
                                                      const int* __restrict__ sl,
                                                      const float* __restrict__ noise,
                                                      float* __restrict__ out) {
    int row = blockIdx.x;                 // 0..ROWS-1
    int b = row >> 11, n = row & 2047;
    int s = sl[row];
    bool last = (n == NN - 1) || (sl[row + 1] != s);
    if (!last) return;
    float nz = noise[row];
    size_t ob = ((size_t)b * NN + s) * DD;
    size_t lb = ((size_t)b * NN + n) * DD;
    int t = threadIdx.x;
    out[ob + t] = H[ob + t] + L[lb + t] * nz;
}

// ---------------------------------------------------------------------------
extern "C" void kernel_launch(void* const* d_in, const int* in_sizes, int n_in,
                              void* d_out, int out_size, void* d_ws, size_t ws_size,
                              hipStream_t stream) {
    const float* LLF = (const float*)d_in[0];
    const float* HLF = (const float*)d_in[1];
    float* out = (float*)d_out;

    char* ws = (char*)d_ws;
    const size_t SZ_BF  = ELEMS * sizeof(unsigned short);   // 16 MB
    const size_t SZ_ROW = (size_t)ROWS * sizeof(float);     // 128 KB
    __bf16* Lb    = (__bf16*)(ws);
    __bf16* Hb    = (__bf16*)(ws + SZ_BF);
    float*  nl    = (float*)(ws + 2 * SZ_BF);
    float*  nh    = (float*)(ws + 2 * SZ_BF + SZ_ROW);
    float*  rnl   = (float*)(ws + 2 * SZ_BF + 2 * SZ_ROW);
    float*  rnh   = (float*)(ws + 2 * SZ_BF + 3 * SZ_ROW);
    int*    idx   = (int*)  (ws + 2 * SZ_BF + 4 * SZ_ROW);
    int*    sl    = (int*)  (ws + 2 * SZ_BF + 5 * SZ_ROW);
    float*  w     = (float*)(ws + 2 * SZ_BF + 6 * SZ_ROW);
    float*  noise = (float*)(ws + 2 * SZ_BF + 7 * SZ_ROW);

    convert_kernel<<<(unsigned)(ELEMS / 256), 256, 0, stream>>>(LLF, HLF, Lb, Hb);
    norm_kernel<<<2 * ROWS, 256, 0, stream>>>(LLF, HLF, nl, nh, rnl, rnh);
    sim_argmax_kernel<<<512, 128, 0, stream>>>(Lb, Hb, rnl, rnh, idx);
    sort_kernel<<<BB, 256, 0, stream>>>(idx, sl);
    w_kernel<<<ROWS / 4, 128, 0, stream>>>(LLF, HLF, nl, nh, sl, w);
    noise_kernel<<<BB, 256, 0, stream>>>(w, noise);
    copy_kernel<<<(unsigned)(ELEMS / 4 / 256), 256, 0, stream>>>((const float4*)HLF,
                                                                 (float4*)out, ELEMS / 4);
    scatter_kernel<<<ROWS, 256, 0, stream>>>(LLF, HLF, sl, noise, out);
}
